// MaxSingularValueLoss_91190745628692
// MI455X (gfx1250) — matrix operand order repaired and checked by reading, not compiled, not run on hardware
//
#include <hip/hip_runtime.h>
#include <hip/hip_bf16.h>

// MI455X (gfx1250): LDS-resident batched Stockham FFT power iteration.
// 512 workgroups (one per (B,I) problem), 256 threads (8 wave32) each.
// Radix-16 FFT stages run the 16-point DFT as v_wmma_f32_16x16x32_f16
// (DFT matrix in the A-operand, [Re;Im]-stacked data in the B-operand),
// radix-2 tail + twiddles in f32 VALU (twiddles via sincos + complex
// recurrence so trans ops stay off the critical path). Spectra are
// precomputed once per launch into d_ws (~115 MB, resident in 192 MB L2).

typedef __attribute__((ext_vector_type(16))) _Float16 v16h;
typedef __attribute__((ext_vector_type(8)))  float    v8f;

#define NFFT     8192
#define NTHREADS 256
#define NPROB    512
#define NITER    100
#define PI_F     3.14159265358979323846f

__device__ __forceinline__ float2 cmulf(float2 a, float2 b) {
    return make_float2(a.x * b.x - a.y * b.y, a.x * b.y + a.y * b.x);
}

// global_prefetch_b8 of a span (each lane's prefetch pulls a full cacheline).
__device__ __forceinline__ void prefetch_span(const float* p, int nfloats, int tid) {
    for (int i = tid * 32; i < nfloats; i += NTHREADS * 32)
        __builtin_prefetch(p + i, 0, 1);
}

// Per-lane constant A-fragments holding the 16x16 DFT matrix in the
// [C | sigma*S] / [-sigma*S | C] real-embedding form (A is 16x32 f16).
// Layout used here, not the ISA table order (16-bit A 16x32): lane M = lane%16;
// half h -> K = h + 16*(lane>=16) (contiguous, matching the B fill).
__device__ __forceinline__ void dft16_frags(int lane, float sigma, v16h& aRe, v16h& aIm) {
    const int k  = lane & 15;
    const int hi = (lane >> 4) * 8;
#pragma unroll
    for (int h = 0; h < 16; ++h) {
        const int v  = h >> 1, lo = h & 1;
        const int K  = h + 2 * hi;
        const int r  = K & 15;
        const float ang = 0.39269908169872415481f * (float)(k * r);  // 2*pi/16
        float s, c;
        __sincosf(ang, &s, &c);
        if (K < 16) { aRe[h] = (_Float16)c;           aIm[h] = (_Float16)(-sigma * s); }
        else        { aRe[h] = (_Float16)(sigma * s); aIm[h] = (_Float16)c; }
    }
}

// One Stockham DIF radix-16 stage: Y[q + s*(16p+k)] = DFT16(a)[k] * w_n^{sigma*p*k},
// a[r] = X[q + s*(n/16)*r + s*p].  Batches 16 (p,q) columns per WMMA pair.
// B layout (32x16 f16): lanes 0-15 carry K=0..15 (Re rows), lanes 16-31 K=16..31 (Im).
__device__ void radix16_stage(const float2* __restrict__ X, float2* __restrict__ Y,
                              int n, int s, int log2s, float sigma,
                              v16h aRe, v16h aIm, int lane, int wave) {
    const int m    = n >> 4;
    const int part = lane >> 4;   // 0: real rows, 1: imag rows
    const int col  = lane & 15;
    const float tw = -6.28318530717958647692f * sigma / (float)n;
    for (int g = wave; g < 32; g += 8) {            // 32 column-groups, 8 waves
        const int f = g * 16 + col;                 // flat (p,q) index
        const int q = f & (s - 1);
        const int p = f >> log2s;
        v16h bFrag;
#pragma unroll
        for (int h = 0; h < 16; ++h) {              // r = h; K = h + 16*part
            const float2 a = X[q + s * (p + m * h)];
            bFrag[h] = (_Float16)(part ? a.y : a.x);
        }
        const v8f zero = {0.f, 0.f, 0.f, 0.f, 0.f, 0.f, 0.f, 0.f};
        v8f dr = __builtin_amdgcn_wmma_f32_16x16x32_f16(false, aRe, false, bFrag,
                                                        (short)0, zero, false, false);
        v8f di = __builtin_amdgcn_wmma_f32_16x16x32_f16(false, aIm, false, bFrag,
                                                        (short)0, zero, false, false);
        // Twiddle w_n^{p*k}, k = d + 8*part: angles are an arithmetic progression
        // in d -> 2 sincos + complex recurrence instead of 8 sincos.
        float wr, wi, css, sns;
        __sincosf(tw * (float)(p * (8 * part)), &wi, &wr);  // base e^{i tw p 8part}
        __sincosf(tw * (float)p, &sns, &css);               // step e^{i tw p}
#pragma unroll
        for (int d = 0; d < 8; ++d) {               // D[k][col], k = d + 8*part
            const int k = d + 8 * part;
            const float yr = dr[d] * wr - di[d] * wi;
            const float yi = dr[d] * wi + di[d] * wr;
            Y[q + s * (16 * p + k)] = make_float2(yr, yi);
            const float nwr = wr * css - wi * sns;
            const float nwi = wr * sns + wi * css;
            wr = nwr; wi = nwi;
        }
    }
}

__device__ void radix2_stage(const float2* __restrict__ X, float2* __restrict__ Y, int tid) {
    for (int q = tid; q < NFFT / 2; q += NTHREADS) {
        const float2 a = X[q], b = X[q + NFFT / 2];
        Y[q]            = make_float2(a.x + b.x, a.y + b.y);
        Y[q + NFFT / 2] = make_float2(a.x - b.x, a.y - b.y);
    }
}

// In: data in A. Out: result in A (stages A->B->A->B->A). sigma=+1 fwd, -1 inv
// (inverse is unscaled; 1/N factors are folded into the precomputed spectra).
__device__ void fft8192(float2* A, float2* B, float sigma, v16h aRe, v16h aIm, int tid) {
    const int lane = tid & 31, wave = tid >> 5;
    radix16_stage(A, B, 8192, 1,   0, sigma, aRe, aIm, lane, wave); __syncthreads();
    radix16_stage(B, A, 512,  16,  4, sigma, aRe, aIm, lane, wave); __syncthreads();
    radix16_stage(A, B, 32,   256, 8, sigma, aRe, aIm, lane, wave); __syncthreads();
    radix2_stage(B, A, tid);                                        __syncthreads();
}

__device__ float block_sum(float v, float* red, int tid) {
    red[tid] = v; __syncthreads();
    for (int off = NTHREADS / 2; off > 0; off >>= 1) {
        if (tid < off) red[tid] += red[tid + off];
        __syncthreads();
    }
    const float r = red[0]; __syncthreads();
    return r;
}

// ws layout per problem (floats): Fc[2*N] | Fte[2*N] | Fto[2*N] | Bv[N]
#define WS_STRIDE ((size_t)(7 * NFFT))

__global__ void msv_setup_kernel(const float* __restrict__ x, const float* __restrict__ circ,
                                 const float* __restrict__ b0, float* __restrict__ ws) {
    extern __shared__ char smem[];
    float2* C1  = (float2*)smem;
    float2* C2  = C1 + NFFT;
    float*  red = (float*)(C2 + NFFT);
    const int pid = blockIdx.x, tid = threadIdx.x;
    const size_t g = (size_t)pid * NFFT;
    float* Fc  = ws + (size_t)pid * WS_STRIDE;
    float* Fte = Fc + 2 * NFFT;
    float* Fto = Fte + 2 * NFFT;
    float* Bv  = Fto + 2 * NFFT;
    v16h aRe, aIm;
    dft16_frags(tid & 31, 1.f, aRe, aIm);
    const float invN = 1.f / (float)NFFT, hInvN = 0.5f / (float)NFFT;
    const float pn = PI_F / (float)NFFT;

    // Fc = FFT(circ) * (1/N)   (1/N = iFFT scaling folded in)
    for (int i = tid; i < NFFT; i += NTHREADS) C1[i] = make_float2(circ[g + i], 0.f);
    __syncthreads();
    fft8192(C1, C2, 1.f, aRe, aIm, tid);
    for (int i = tid; i < NFFT; i += NTHREADS) { Fc[2*i] = C1[i].x * invN; Fc[2*i+1] = C1[i].y * invN; }
    __syncthreads();

    // Toeplitz circulant embedding c2 = [a0..a_{N-1}, 0, a_{N-1}..a1] (len 2N).
    // Even bins:  Fte[k] = FFT_N(c_lo + c_hi)[k] * (1/2N)
    for (int i = tid; i < NFFT; i += NTHREADS) {
        const float ch = (i == 0) ? 0.f : x[g + NFFT - i];
        C1[i] = make_float2(x[g + i] + ch, 0.f);
    }
    __syncthreads();
    fft8192(C1, C2, 1.f, aRe, aIm, tid);
    for (int i = tid; i < NFFT; i += NTHREADS) { Fte[2*i] = C1[i].x * hInvN; Fte[2*i+1] = C1[i].y * hInvN; }
    __syncthreads();

    // Odd bins:  Fto[k] = FFT_N((c_lo - c_hi) .* e^{-i*pi*n/N})[k] * (1/2N)
    for (int i = tid; i < NFFT; i += NTHREADS) {
        const float ch = (i == 0) ? 0.f : x[g + NFFT - i];
        const float d  = x[g + i] - ch;
        float sn, cs;
        __sincosf(-pn * (float)i, &sn, &cs);
        C1[i] = make_float2(d * cs, d * sn);
    }
    __syncthreads();
    fft8192(C1, C2, 1.f, aRe, aIm, tid);
    for (int i = tid; i < NFFT; i += NTHREADS) { Fto[2*i] = C1[i].x * hInvN; Fto[2*i+1] = C1[i].y * hInvN; }
    __syncthreads();

    // Bv = b0 / ||b0||
    float ss = 0.f;
    for (int i = tid; i < NFFT; i += NTHREADS) { const float v = b0[g + i]; ss += v * v; }
    ss = block_sum(ss, red, tid);
    const float rn = rsqrtf(ss);
    for (int i = tid; i < NFFT; i += NTHREADS) Bv[i] = b0[g + i] * rn;
}

__global__ void msv_iterate_kernel(const float* __restrict__ ws, float* __restrict__ out) {
    extern __shared__ char smem[];
    float2* C1  = (float2*)smem;          // ping
    float2* C2  = C1 + NFFT;              // pong
    float2* H   = C2 + NFFT;              // holds Bhat*Fc*Fte -> u
    float*  Br  = (float*)(H + NFFT);     // current b (real)
    float*  red = Br + NFFT;
    const int pid = blockIdx.x, tid = threadIdx.x;
    const float* Fc  = ws + (size_t)pid * WS_STRIDE;
    const float* Fte = Fc + 2 * NFFT;
    const float* Fto = Fte + 2 * NFFT;
    const float* Bv  = Fto + 2 * NFFT;
    v16h aReF, aImF, aReI, aImI;
    dft16_frags(tid & 31,  1.f, aReF, aImF);
    dft16_frags(tid & 31, -1.f, aReI, aImI);
    const float pn = PI_F / (float)NFFT;

    for (int i = tid; i < NFFT; i += NTHREADS) Br[i] = Bv[i];
    __syncthreads();

    for (int it = 0; it <= NITER; ++it) {
        // Warm the WGP path for this iteration's spectra (L2-resident).
        prefetch_span(Fc,  2 * NFFT, tid);
        prefetch_span(Fte, 2 * NFFT, tid);
        prefetch_span(Fto, 2 * NFFT, tid);

        // ---- Bhat = FFT(b)
        for (int i = tid; i < NFFT; i += NTHREADS) C1[i] = make_float2(Br[i], 0.f);
        __syncthreads();
        fft8192(C1, C2, 1.f, aReF, aImF, tid);

        // ---- Even-bin branch folded in the spectral domain:
        //      E = Bhat.*Fc (circ_mul spectrum) ; H = E.*Fte (even bins * spectrum)
        for (int i = tid; i < NFFT; i += NTHREADS) {
            const float2 e = cmulf(C1[i], make_float2(Fc[2*i], Fc[2*i+1]));
            H[i]  = cmulf(e, make_float2(Fte[2*i], Fte[2*i+1]));
            C1[i] = e;
        }
        __syncthreads();

        // ---- y = Re(iFFT(E)); odd-bin branch needs time-domain modulation
        fft8192(C1, C2, -1.f, aReI, aImI, tid);
        for (int i = tid; i < NFFT; i += NTHREADS) {
            float sn, cs;
            __sincosf(-pn * (float)i, &sn, &cs);       // mu_n = e^{-i pi n/N}
            const float yv = C1[i].x;
            C1[i] = make_float2(yv * cs, yv * sn);
        }
        __syncthreads();
        fft8192(C1, C2, 1.f, aReF, aImF, tid);
        for (int i = tid; i < NFFT; i += NTHREADS)
            C1[i] = cmulf(C1[i], make_float2(Fto[2*i], Fto[2*i+1]));
        __syncthreads();
        fft8192(C1, C2, -1.f, aReI, aImI, tid);   // v in C1
        fft8192(H,  C2, -1.f, aReI, aImI, tid);   // u in H (C1 untouched)

        // ---- t[n] = Re(u[n]) + Re(e^{+i pi n/N} v[n])   (scalings pre-folded)
        if (it < NITER) {
            float ss = 0.f;
            for (int i = tid; i < NFFT; i += NTHREADS) {
                float sn, cs;
                __sincosf(pn * (float)i, &sn, &cs);
                const float t  = H[i].x + cs * C1[i].x - sn * C1[i].y;
                const float bn = Br[i] - t;
                Br[i] = bn;
                ss += bn * bn;
            }
            ss = block_sum(ss, red, tid);
            const float rn = rsqrtf(ss);
            for (int i = tid; i < NFFT; i += NTHREADS) Br[i] *= rn;
            __syncthreads();
        } else {
            float sg = 0.f;
            for (int i = tid; i < NFFT; i += NTHREADS) {
                float sn, cs;
                __sincosf(pn * (float)i, &sn, &cs);
                const float t = H[i].x + cs * C1[i].x - sn * C1[i].y;
                sg += Br[i] * (Br[i] - t);
            }
            sg = block_sum(sg, red, tid);
            if (tid == 0) atomicAdd(out, fabsf(sg) * (1.f / (float)NPROB));
        }
    }
}

extern "C" void kernel_launch(void* const* d_in, const int* in_sizes, int n_in,
                              void* d_out, int out_size, void* d_ws, size_t ws_size,
                              hipStream_t stream) {
    (void)in_sizes; (void)n_in; (void)out_size; (void)ws_size;
    const float* x    = (const float*)d_in[0];
    const float* circ = (const float*)d_in[1];
    const float* b0   = (const float*)d_in[2];
    float* ws  = (float*)d_ws;   // needs 512 * 7 * 8192 * 4 B ~= 115 MB
    float* out = (float*)d_out;

    const size_t ldsSetup = (size_t)(2 * NFFT) * sizeof(float2) + NTHREADS * sizeof(float);
    const size_t ldsIter  = (size_t)(3 * NFFT) * sizeof(float2)
                          + (size_t)(NFFT + NTHREADS) * sizeof(float);  // ~225 KB < 320 KB/WGP

    hipFuncSetAttribute(reinterpret_cast<const void*>(msv_setup_kernel),
                        hipFuncAttributeMaxDynamicSharedMemorySize, (int)ldsSetup);
    hipFuncSetAttribute(reinterpret_cast<const void*>(msv_iterate_kernel),
                        hipFuncAttributeMaxDynamicSharedMemorySize, (int)ldsIter);

    hipMemsetAsync(d_out, 0, sizeof(float), stream);
    msv_setup_kernel<<<NPROB, NTHREADS, ldsSetup, stream>>>(x, circ, b0, ws);
    msv_iterate_kernel<<<NPROB, NTHREADS, ldsIter, stream>>>(ws, out);
}
